// Attention_42949673623
// MI455X (gfx1250) — matrix operand order repaired and checked by reading, not compiled, not run on hardware
//
#include <hip/hip_runtime.h>
#include <hip/hip_bf16.h>

// ---------------------------------------------------------------------------
// Types for WMMA fragments (CDNA5 gfx1250, wave32)
// ---------------------------------------------------------------------------
typedef __attribute__((ext_vector_type(16))) _Float16 v16h;
typedef __attribute__((ext_vector_type(8)))  _Float16 v8h;
typedef __attribute__((ext_vector_type(8)))  float    v8f;
typedef __attribute__((ext_vector_type(4)))  float    v4f;
typedef __attribute__((ext_vector_type(4)))  unsigned u4v;
typedef __attribute__((ext_vector_type(8)))  unsigned u8v;

__device__ __forceinline__ v8f wmma_f16(v16h a, v16h b, v8f c) {
    // D = A(16x32 f16) * B(32x16 f16) + C(16x16 f32)
    return __builtin_amdgcn_wmma_f32_16x16x32_f16(
        /*neg_a=*/false, a, /*neg_b=*/false, b,
        /*c_mod=*/(short)0, c, /*reuse_a=*/false, /*reuse_b=*/false);
}

#define EMBED 256
#define HEADS 8
#define HW    32
#define BATCH 2
#define SEQ   2048

// ---------------------------------------------------------------------------
// TDM: issue a 2D tensor_load_to_lds for one bias slab:
//   tile = 16 rows x 256 f32 (16 q-rows x (32 kv * 8 heads)), row stride
//   SEQ*HEADS elements.  Descriptor per CDNA5 ISA 8.3/8.4 (groups 0 and 1,
//   VADDR2/VADDR3 omitted for a 2D tensor).  Tracked by TENSORcnt.
// ---------------------------------------------------------------------------
__device__ __forceinline__ void tdm_load_bias_slab(const float* gptr,
                                                   unsigned lds_byte_addr) {
    const unsigned long long ga = (unsigned long long)(uintptr_t)gptr;
    u4v g0;
    g0[0] = 1u;                                  // count=1 (valid user D#)
    g0[1] = lds_byte_addr;                       // lds_addr (bytes)
    g0[2] = (unsigned)(ga & 0xffffffffu);        // global_addr[31:0]
    g0[3] = (unsigned)((ga >> 32) & 0x01ffffffu) // global_addr[56:32]
          | (2u << 30);                          // type = 2 ("image")
    u8v g1;
    g1[0] = 2u << 16;                            // data_size = 2 -> 4 bytes
    g1[1] = (16384u & 0xffffu) << 16;            // tensor_dim0 lo16 @ [31:16]
    g1[2] = (16384u >> 16) | (16u << 16);        // tensor_dim0 hi16 | tensor_dim1 lo16
    g1[3] = (256u << 16);                        // tensor_dim1 hi16 | tile_dim0=256
    g1[4] = 16u;                                 // tile_dim1=16, tile_dim2=0
    g1[5] = 16384u;                              // tensor_dim0_stride lo32 (= SEQ*HEADS)
    g1[6] = 0u;                                  // stride0 hi16 | stride1 lo16
    g1[7] = 0u;                                  // stride1 hi32
    asm volatile("tensor_load_to_lds %0, %1" :: "s"(g0), "s"(g1) : "memory");
}

// ===========================================================================
// Kernel 1: fused QKV projection.
//   t[m][f] = sum_e x[m][e] * Wp[f][e]   (M=B*L=4096, F=768, K=256)
// Per 96-wide head block f -> [q(32) k(32) v(32)].
// Writes: Qh (B,H,L,32) f16 pre-scaled by hw^-0.5, Kh (B,H,L,32) f16,
//         VTh (B,H,32,L) f16  (transposed so PV B-frags are contiguous).
// One wave computes one 16x16 tile; 8 waves / block.
// ===========================================================================
__global__ __launch_bounds__(256) void qkv_proj_kernel(
    const float* __restrict__ x, const float* __restrict__ Wp,
    _Float16* __restrict__ Qh, _Float16* __restrict__ Kh,
    _Float16* __restrict__ VTh)
{
    const int lane = threadIdx.x & 31;
    const int wave = threadIdx.x >> 5;
    const int tile = blockIdx.x * 8 + wave;       // 0..12287
    const int tM   = tile / 48;                   // 0..255 (row tiles of 4096)
    const int tF   = tile % 48;                   // 0..47  (col tiles of 768)
    const int hi   = lane >> 4;
    const int lo   = lane & 15;

    const int m = tM * 16 + lo;                   // A-frag row
    const int f = tF * 16 + lo;                   // B-frag col
    const float* xrow = x  + (size_t)m * EMBED;
    const float* wrow = Wp + (size_t)f * EMBED;

    v8f acc = {};
    #pragma unroll
    for (int e0 = 0; e0 < EMBED; e0 += 32) {
        v16h a, bfr;
        // A-frag: lane holds row m, K runs {e0+hi*8..+7} and {e0+16+hi*8..+7}
        const float* p0 = xrow + e0 + hi * 8;
        #pragma unroll
        for (int j = 0; j < 8; ++j) {
            a[j]     = (_Float16)p0[j];
            a[8 + j] = (_Float16)p0[16 + j];
        }
        // B-frag: lane holds col f, K runs {e0+hi*8..+7} and {e0+16+hi*8..+7}
        const float* p1 = wrow + e0 + hi * 8;
        #pragma unroll
        for (int j = 0; j < 8; ++j) { bfr[j] = (_Float16)p1[j]; bfr[8 + j] = (_Float16)p1[16 + j]; }
        acc = wmma_f16(a, bfr, acc);
    }

    // D layout: row = tM*16 + r + 8*hi, col = f. Scatter into Q/K/VT.
    const int hh  = f / 96;
    const int r96 = f % 96;     // 16-tile never crosses q/k/v boundaries
    const float qscale = 0.17677669529663687f;    // 32^-0.5
    #pragma unroll
    for (int r = 0; r < 8; ++r) {
        const int mr = tM * 16 + r + 8 * hi;
        const int b  = mr >> 11;
        const int l  = mr & (SEQ - 1);
        const float val = acc[r];
        const size_t head = (size_t)(b * HEADS + hh);
        if (r96 < 32) {
            Qh[(head * SEQ + l) * HW + r96] = (_Float16)(val * qscale);
        } else if (r96 < 64) {
            Kh[(head * SEQ + l) * HW + (r96 - 32)] = (_Float16)val;
        } else {
            VTh[(head * HW + (r96 - 64)) * SEQ + l] = (_Float16)val;
        }
    }
}

// ===========================================================================
// Kernel 2: flash-attention with additive bias.
// Block = (b, q-tile of 16). 8 waves = 8 heads. The 16x32x8-f32 bias slab is
// DMA'd global->LDS by the Tensor Data Mover (double-buffered: TDM for slab
// i+1 overlaps compute on slab i). Online softmax in f32, matmuls in f16
// WMMA. P converted D-layout -> A-layout via per-wave LDS.
// Output y: (B, L, H*HW) f32.
// ===========================================================================
__global__ __launch_bounds__(256) void attn_kernel(
    const _Float16* __restrict__ Qh, const _Float16* __restrict__ Kh,
    const _Float16* __restrict__ VTh, const float* __restrict__ bias,
    float* __restrict__ y)
{
    __shared__ __align__(16) float    biasS[2][16 * 32 * HEADS]; // 2 x 16KB
    __shared__ __align__(16) _Float16 pS[8][16 * 32];            // 8KB

    const int tid  = threadIdx.x;
    const int lane = tid & 31;
    const int wave = tid >> 5;          // head index
    const int b    = blockIdx.x >> 7;   // / (SEQ/16)
    const int qt   = blockIdx.x & 127;
    const int hi   = lane >> 4;
    const int lo   = lane & 15;
    const int h    = wave;

    const size_t headBase = (size_t)(b * HEADS + h) * SEQ;

    // Persistent Q A-frag for this wave's 16 q-rows.
    v16h aq;
    {
        const _Float16* qp = Qh + (headBase + qt * 16 + lo) * HW;
        const v8h q0 = *(const v8h*)(qp + hi * 8);
        const v8h q1 = *(const v8h*)(qp + 16 + hi * 8);
        #pragma unroll
        for (int j = 0; j < 8; ++j) { aq[j] = q0[j]; aq[8 + j] = q1[j]; }
    }

    v8f o0 = {}, o1 = {};
    float mrow[8], lrow[8];
    #pragma unroll
    for (int r = 0; r < 8; ++r) { mrow[r] = -1e30f; lrow[r] = 0.0f; }

    const float* biasBase = bias + ((size_t)(b * SEQ + qt * 16)) * SEQ * HEADS;
    const _Float16* vhead = VTh + (size_t)(b * HEADS + h) * HW * SEQ;

    // Prologue: TDM DMA of slab 0 into buffer 0 (one issue per block).
    if (wave == 0) {
        tdm_load_bias_slab(biasBase, (unsigned)(uintptr_t)&biasS[0][0]);
    }

    for (int it = 0; it < SEQ / 32; ++it) {
        const int kv0 = it * 32;
        const int buf = it & 1;

        if (wave == 0) {
            asm volatile("s_wait_tensorcnt 0x0" ::: "memory");
        }
        __syncthreads();   // (A) slab `it` visible to all waves

        // Kick off DMA for slab it+1 into the other buffer; its target was
        // last read in iteration it-1, which completed before barrier (A).
        if (wave == 0 && it + 1 < SEQ / 32) {
            tdm_load_bias_slab(biasBase + (size_t)(it + 1) * 32 * HEADS,
                               (unsigned)(uintptr_t)&biasS[buf ^ 1][0]);
        }
        const float* bS = biasS[buf];

        // --- K B-frags (two 16-wide kv sub-tiles): K runs {hi*8..+7} and {16+hi*8..+7}
        v16h bk0, bk1, bv0, bv1;
        {
            const _Float16* kp0 = Kh + (headBase + kv0 + lo) * HW + hi * 8;
            const _Float16* kp1 = Kh + (headBase + kv0 + 16 + lo) * HW + hi * 8;
            const v8h t0 = *(const v8h*)kp0, t1 = *(const v8h*)(kp0 + 16);
            const v8h t2 = *(const v8h*)kp1, t3 = *(const v8h*)(kp1 + 16);
            #pragma unroll
            for (int j = 0; j < 8; ++j) {
                bk0[j] = t0[j]; bk0[8 + j] = t1[j];
                bk1[j] = t2[j]; bk1[8 + j] = t3[j];
            }
        }
        // --- V B-frags (two channel halves) from transposed V: contiguous
        {
            const _Float16* vp0 = vhead + (size_t)lo * SEQ        + kv0 + hi * 8;
            const _Float16* vp1 = vhead + (size_t)(16 + lo) * SEQ + kv0 + hi * 8;
            const v8h t0 = *(const v8h*)vp0, t1 = *(const v8h*)(vp0 + 16);
            const v8h t2 = *(const v8h*)vp1, t3 = *(const v8h*)(vp1 + 16);
            #pragma unroll
            for (int j = 0; j < 8; ++j) {
                bv0[j] = t0[j]; bv0[8 + j] = t1[j];
                bv1[j] = t2[j]; bv1[8 + j] = t3[j];
            }
        }

        // --- S = Q K^T  (two 16x16 f32 tiles covering 32 kv)
        v8f zero = {};
        v8f s0 = wmma_f16(aq, bk0, zero);
        v8f s1 = wmma_f16(aq, bk1, zero);

        // --- bias add + online softmax (D layout: row r+8*hi, col lo)
        float pmax[8];
        #pragma unroll
        for (int r = 0; r < 8; ++r) {
            const int qr = r + 8 * hi;
            s0[r] += bS[(qr * 32 + lo) * HEADS + h];
            s1[r] += bS[(qr * 32 + 16 + lo) * HEADS + h];
            pmax[r] = fmaxf(s0[r], s1[r]);
        }
        #pragma unroll
        for (int msk = 1; msk < 16; msk <<= 1) {
            #pragma unroll
            for (int r = 0; r < 8; ++r)
                pmax[r] = fmaxf(pmax[r], __shfl_xor(pmax[r], msk, 32));
        }
        float psum[8];
        #pragma unroll
        for (int r = 0; r < 8; ++r) {
            const float mnew = fmaxf(mrow[r], pmax[r]);
            const float corr = __expf(mrow[r] - mnew);
            mrow[r] = mnew;
            s0[r] = __expf(s0[r] - mnew);
            s1[r] = __expf(s1[r] - mnew);
            o0[r] *= corr; o1[r] *= corr;
            lrow[r] *= corr;
            psum[r] = s0[r] + s1[r];
        }
        #pragma unroll
        for (int msk = 1; msk < 16; msk <<= 1) {
            #pragma unroll
            for (int r = 0; r < 8; ++r)
                psum[r] += __shfl_xor(psum[r], msk, 32);
        }
        #pragma unroll
        for (int r = 0; r < 8; ++r) lrow[r] += psum[r];

        // --- stage P (f16, 16x32 row-major) through per-wave LDS -> A-frag
        _Float16* pw = pS[wave];
        #pragma unroll
        for (int r = 0; r < 8; ++r) {
            const int qr = r + 8 * hi;
            pw[qr * 32 + lo]      = (_Float16)s0[r];
            pw[qr * 32 + 16 + lo] = (_Float16)s1[r];
        }
        asm volatile("s_wait_dscnt 0" ::: "memory");
        v16h ap;
        {
            const _Float16* pr = pw + lo * 32;
            const v8h p0 = *(const v8h*)(pr + hi * 8);
            const v8h p1 = *(const v8h*)(pr + 16 + hi * 8);
            #pragma unroll
            for (int j = 0; j < 8; ++j) { ap[j] = p0[j]; ap[8 + j] = p1[j]; }
        }

        // --- O += P V  (two channel halves)
        o0 = wmma_f16(ap, bv0, o0);
        o1 = wmma_f16(ap, bv1, o1);

        __syncthreads();   // (B) all waves done reading buf before reuse
    }

    // Epilogue: normalize and write y (B, L, H*HW) f32.
    #pragma unroll
    for (int r = 0; r < 8; ++r) {
        const int qr = r + 8 * hi;
        const float inv = 1.0f / lrow[r];
        const int q = qt * 16 + qr;
        float* yp = y + ((size_t)(b * SEQ + q)) * EMBED + h * HW;
        yp[lo]      = o0[r] * inv;
        yp[16 + lo] = o1[r] * inv;
    }
}

// ===========================================================================
// Kernel 3: output projection. out[m][f] = sum_e y[m][e]*Wo[f][e] + bo[f]
// M=4096, N=256, K=256. Same WMMA tiling as kernel 1.
// ===========================================================================
__global__ __launch_bounds__(256) void out_proj_kernel(
    const float* __restrict__ y, const float* __restrict__ Wo,
    const float* __restrict__ bo, float* __restrict__ out)
{
    const int lane = threadIdx.x & 31;
    const int wave = threadIdx.x >> 5;
    const int tile = blockIdx.x * 8 + wave;       // 0..4095
    const int tM   = tile >> 4;                   // 0..255
    const int tN   = tile & 15;                   // 0..15
    const int hi   = lane >> 4;
    const int lo   = lane & 15;

    const int m = tM * 16 + lo;
    const int f = tN * 16 + lo;
    const float* yrow = y  + (size_t)m * EMBED;
    const float* wrow = Wo + (size_t)f * EMBED;

    v8f acc = {};
    #pragma unroll
    for (int e0 = 0; e0 < EMBED; e0 += 32) {
        v16h a, bfr;
        const float* p0 = yrow + e0 + hi * 8;
        #pragma unroll
        for (int j = 0; j < 8; ++j) {
            a[j]     = (_Float16)p0[j];
            a[8 + j] = (_Float16)p0[16 + j];
        }
        const float* p1 = wrow + e0 + hi * 8;
        #pragma unroll
        for (int j = 0; j < 8; ++j) { bfr[j] = (_Float16)p1[j]; bfr[8 + j] = (_Float16)p1[16 + j]; }
        acc = wmma_f16(a, bfr, acc);
    }

    const float bias_o = bo[f];
    #pragma unroll
    for (int r = 0; r < 8; ++r) {
        const int mr = tM * 16 + r + 8 * hi;
        out[(size_t)mr * EMBED + f] = acc[r] + bias_o;
    }
}

// ===========================================================================
// Host launcher
// ===========================================================================
extern "C" void kernel_launch(void* const* d_in, const int* in_sizes, int n_in,
                              void* d_out, int out_size, void* d_ws, size_t ws_size,
                              hipStream_t stream) {
    (void)in_sizes; (void)n_in; (void)out_size; (void)ws_size;
    const float* x    = (const float*)d_in[0];   // (B, L, E)
    const float* bias = (const float*)d_in[1];   // (B, L, L, H)
    const float* Wp   = (const float*)d_in[2];   // (3E, E)
    const float* Wo   = (const float*)d_in[3];   // (E, E)
    const float* bo   = (const float*)d_in[4];   // (E,)
    float* out = (float*)d_out;                  // (B, L, E)

    char* ws = (char*)d_ws;
    _Float16* Qh  = (_Float16*)(ws);                       // 2 MB
    _Float16* Kh  = (_Float16*)(ws + (size_t)(2u << 20));  // 2 MB
    _Float16* VTh = (_Float16*)(ws + (size_t)(4u << 20));  // 2 MB
    float*    yb  = (float*)   (ws + (size_t)(6u << 20));  // 4 MB

    // 12288 (M,F) tiles / 8 waves per block
    qkv_proj_kernel<<<1536, 256, 0, stream>>>(x, Wp, Qh, Kh, VTh);
    // B * (L/16) blocks, 8 heads per block
    attn_kernel<<<BATCH * (SEQ / 16), 256, 0, stream>>>(Qh, Kh, VTh, bias, yb);
    // 4096 (M,N) tiles / 8 waves per block
    out_proj_kernel<<<512, 256, 0, stream>>>(yb, Wo, bo, out);
}
